// GAT_LSTM_34840774705238
// MI455X (gfx1250) — matrix operand order repaired and checked by reading, not compiled, not run on hardware
//
#include <hip/hip_runtime.h>
#include <math.h>

#define NND 10000
#define TT 24
#define IND 16
#define HIDD 128
#define NHEADS 4
#define NE 160000
#define PREDL 24

typedef __bf16 bf16_t;
typedef __attribute__((ext_vector_type(8)))  __bf16 v8bf;
typedef __attribute__((ext_vector_type(16))) __bf16 v16bf;
typedef __attribute__((ext_vector_type(8)))  float  v8f;

// ---------------------------------------------------------------------------
// WMMA GEMM:  C[M,Ncols] = act( A[M,K] @ W[Ncols,K]^T  (+ A2 @ W2^T) + b1 + b2 )
// Per-wave 16x64 C tile (4x 16x16 accumulators): one A fragment is reused
// across 4 WMMAs per K-step -> 10 b128 loads / 4 WMMAs instead of 16 / 4.
// A row-major bf16 with row stride lda and element offset aoff.
// W row-major [Ncols_padded64rows, K] bf16 (PyTorch weight layout: out x in).
// Fragment layouts used here (wave32; contiguous k, not the ISA table order):
//   A lane l: row = l&15; halves 0-15 = K[(l>>4)*16 .. +15]
//   B lane l: col = l&15; halves 0-15 = K[(l>>4)*16 .. +15]  (contiguous from W row)
//   D lane l, vgpr v: row = v + 8*(l>>4), col = l&15
// ---------------------------------------------------------------------------
__device__ __forceinline__ v16bf load_a_frag(const bf16_t* __restrict__ arow, int ks) {
  v8bf lo = *(const v8bf*)(arow + ks * 16);
  v8bf hi = *(const v8bf*)(arow + ks * 16 + 8);
  return __builtin_shufflevector(lo, hi, 0,1,2,3,4,5,6,7,8,9,10,11,12,13,14,15);
}

template <typename OutT, int ACT>
__global__ void wmma_gemm_kernel(const bf16_t* __restrict__ A, long lda, long aoff,
                                 const bf16_t* __restrict__ W,
                                 const bf16_t* __restrict__ A2, long lda2,
                                 const bf16_t* __restrict__ W2,
                                 const float* __restrict__ bias1,
                                 const float* __restrict__ bias2,
                                 OutT* __restrict__ C, long ldc,
                                 int M, int Ncols, int K, int K2) {
  const int wave   = (int)((blockIdx.x * (unsigned)blockDim.x + threadIdx.x) >> 5);
  const int lane   = threadIdx.x & 31;
  const int tilesN = (Ncols + 63) >> 6;
  const int tM = wave / tilesN;
  const int tN = wave - tM * tilesN;
  if (tM * 16 >= M) return;                 // wave-uniform exit (EXEC all-1 for WMMA)

  const int ar = lane & 15;
  const int ks = lane >> 4;
  v8f acc[4] = {{}, {}, {}, {}};
  {
    const bf16_t* arow  = A + aoff + (long)(tM * 16 + ar) * lda;
    const bf16_t* wbase = W + (long)(tN * 64 + ar) * K + ks * 16;
    for (int k = 0; k < K; k += 32) {
      v16bf a  = load_a_frag(arow + k, ks);
      v16bf b0 = *(const v16bf*)(wbase + k);
      v16bf b1 = *(const v16bf*)(wbase + (long)16 * K + k);
      v16bf b2 = *(const v16bf*)(wbase + (long)32 * K + k);
      v16bf b3 = *(const v16bf*)(wbase + (long)48 * K + k);
      acc[0] = __builtin_amdgcn_wmma_f32_16x16x32_bf16(false, a, false, b0, (short)0, acc[0], false, false);
      acc[1] = __builtin_amdgcn_wmma_f32_16x16x32_bf16(false, a, false, b1, (short)0, acc[1], false, false);
      acc[2] = __builtin_amdgcn_wmma_f32_16x16x32_bf16(false, a, false, b2, (short)0, acc[2], false, false);
      acc[3] = __builtin_amdgcn_wmma_f32_16x16x32_bf16(false, a, false, b3, (short)0, acc[3], false, false);
    }
  }
  if (A2 != nullptr) {                      // fused second GEMM (LSTM recurrent term)
    const bf16_t* arow  = A2 + (long)(tM * 16 + ar) * lda2;
    const bf16_t* wbase = W2 + (long)(tN * 64 + ar) * K2 + ks * 16;
    for (int k = 0; k < K2; k += 32) {
      v16bf a  = load_a_frag(arow + k, ks);
      v16bf b0 = *(const v16bf*)(wbase + k);
      v16bf b1 = *(const v16bf*)(wbase + (long)16 * K2 + k);
      v16bf b2 = *(const v16bf*)(wbase + (long)32 * K2 + k);
      v16bf b3 = *(const v16bf*)(wbase + (long)48 * K2 + k);
      acc[0] = __builtin_amdgcn_wmma_f32_16x16x32_bf16(false, a, false, b0, (short)0, acc[0], false, false);
      acc[1] = __builtin_amdgcn_wmma_f32_16x16x32_bf16(false, a, false, b1, (short)0, acc[1], false, false);
      acc[2] = __builtin_amdgcn_wmma_f32_16x16x32_bf16(false, a, false, b2, (short)0, acc[2], false, false);
      acc[3] = __builtin_amdgcn_wmma_f32_16x16x32_bf16(false, a, false, b3, (short)0, acc[3], false, false);
    }
  }
  const int rbase = tM * 16 + (ks << 3);
#pragma unroll
  for (int s = 0; s < 4; ++s) {
    const int col = tN * 64 + s * 16 + ar;
    if (col < Ncols) {
      float bv = 0.f;
      if (bias1) bv += bias1[col];
      if (bias2) bv += bias2[col];
#pragma unroll
      for (int v = 0; v < 8; ++v) {
        float x = acc[s][v] + bv;
        if (ACT == 1) x = x > 0.f ? x : 0.f;
        C[(long)(rbase + v) * ldc + col] = (OutT)x;
      }
    }
  }
}

// ---------------------------------------------------------------------------
// Elementwise kernels
// ---------------------------------------------------------------------------
__global__ void cast_kernel(const float* __restrict__ in, bf16_t* __restrict__ out, int n) {
  int i = blockIdx.x * blockDim.x + threadIdx.x;
  if (i < n) out[i] = (bf16_t)in[i];
}

__global__ void cast_pad2d_kernel(const float* __restrict__ in, bf16_t* __restrict__ out,
                                  int irows, int icols, int orows, int ocols) {
  int i = blockIdx.x * blockDim.x + threadIdx.x;
  int tot = orows * ocols;
  if (i >= tot) return;
  int r = i / ocols, c = i - r * ocols;
  out[i] = (r < irows && c < icols) ? (bf16_t)in[r * icols + c] : (bf16_t)0.f;
}

__global__ void fill_kernel(float* __restrict__ p, float v, int n) {
  int i = blockIdx.x * blockDim.x + threadIdx.x;
  if (i < n) p[i] = v;
}

__device__ __forceinline__ float sigm(float x) { return 1.f / (1.f + expf(-x)); }

__global__ void lstm_cell_kernel(const float* __restrict__ g, float* __restrict__ c,
                                 bf16_t* __restrict__ h, bf16_t* __restrict__ hseq, int t) {
  int i = blockIdx.x * blockDim.x + threadIdx.x;
  if (i >= NND * HIDD) return;
  int n = i >> 7, j = i & 127;
  const float* gr = g + (long)n * (4 * HIDD);
  float ig = sigm(gr[j]);
  float fg = sigm(gr[HIDD + j]);
  float gg = tanhf(gr[2 * HIDD + j]);
  float og = sigm(gr[3 * HIDD + j]);
  float cc = fg * c[i] + ig * gg;
  c[i] = cc;
  float hh = og * tanhf(cc);
  h[i] = (bf16_t)hh;
  if (hseq) hseq[((long)n * TT + t) * HIDD + j] = (bf16_t)hh;
}

__device__ __forceinline__ void atomicMaxFloat(float* addr, float val) {
  if (val >= 0.f) atomicMax((int*)addr, __float_as_int(val));
  else            atomicMin((unsigned int*)addr, (unsigned int)__float_as_int(val));
}

__global__ void gat_score_kernel(const int* __restrict__ ei, const float* __restrict__ xl,
                                 const float* __restrict__ xr, const float* __restrict__ att,
                                 float* __restrict__ score, float* __restrict__ smax) {
  int t = blockIdx.x * blockDim.x + threadIdx.x;
  if (t >= NE * NHEADS) return;
  int e = t >> 2, h = t & 3;
  int s = ei[e], d = ei[NE + e];
  const float4* pl = (const float4*)(xl + (long)s * 512 + h * 128);
  const float4* pr = (const float4*)(xr + (long)d * 512 + h * 128);
  const float4* pa = (const float4*)(att + h * 128);
  float acc = 0.f;
#pragma unroll 4
  for (int c = 0; c < 32; ++c) {
    float4 a = pl[c], b = pr[c], w = pa[c];
    float v;
    v = a.x + b.x; v = v > 0.f ? v : 0.2f * v; acc += w.x * v;
    v = a.y + b.y; v = v > 0.f ? v : 0.2f * v; acc += w.y * v;
    v = a.z + b.z; v = v > 0.f ? v : 0.2f * v; acc += w.z * v;
    v = a.w + b.w; v = v > 0.f ? v : 0.2f * v; acc += w.w * v;
  }
  score[t] = acc;
  atomicMaxFloat(&smax[d * NHEADS + h], acc);
}

__global__ void gat_norm_kernel(const int* __restrict__ ei, float* __restrict__ score,
                                const float* __restrict__ smax, float* __restrict__ denom) {
  int t = blockIdx.x * blockDim.x + threadIdx.x;
  if (t >= NE * NHEADS) return;
  int e = t >> 2, h = t & 3;
  int d = ei[NE + e];
  float p = expf(score[t] - smax[d * NHEADS + h]);
  score[t] = p;
  atomicAdd(&denom[d * NHEADS + h], p);
}

__global__ void gat_agg_kernel(const int* __restrict__ ei, const float* __restrict__ p,
                               const float* __restrict__ denom, const float* __restrict__ xl,
                               float* __restrict__ agg) {
  long t = (long)blockIdx.x * blockDim.x + threadIdx.x;
  if (t >= (long)NE * 512) return;
  int e = (int)(t >> 9);
  int rem = (int)(t & 511);
  int h = rem >> 7;
  int s = ei[e], d = ei[NE + e];
  float w = p[(long)e * NHEADS + h] / (denom[d * NHEADS + h] + 1e-16f);
  atomicAdd(&agg[(long)d * 512 + rem], w * xl[(long)s * 512 + rem]);
}

__global__ void addbias_cast_kernel(const float* __restrict__ a, const float* __restrict__ b,
                                    bf16_t* __restrict__ o, int n) {
  int i = blockIdx.x * blockDim.x + threadIdx.x;
  if (i < n) o[i] = (bf16_t)(a[i] + b[i & 511]);
}

// ---------------------------------------------------------------------------
// Host orchestration
// ---------------------------------------------------------------------------
static void launch_gemm(hipStream_t st, int outBf16, int act,
                        const bf16_t* A, long lda, long aoff, const bf16_t* W,
                        const bf16_t* A2, long lda2, const bf16_t* W2,
                        const float* b1, const float* b2,
                        void* C, long ldc, int M, int Nc, int K, int K2) {
  int tilesN = (Nc + 63) >> 6;
  long waves = (long)((M + 15) >> 4) * tilesN;
  long blocks = (waves * 32 + 255) / 256;
  if (!outBf16) {
    wmma_gemm_kernel<float, 0><<<dim3((unsigned)blocks), dim3(256), 0, st>>>(
        A, lda, aoff, W, A2, lda2, W2, b1, b2, (float*)C, ldc, M, Nc, K, K2);
  } else if (act == 0) {
    wmma_gemm_kernel<bf16_t, 0><<<dim3((unsigned)blocks), dim3(256), 0, st>>>(
        A, lda, aoff, W, A2, lda2, W2, b1, b2, (bf16_t*)C, ldc, M, Nc, K, K2);
  } else {
    wmma_gemm_kernel<bf16_t, 1><<<dim3((unsigned)blocks), dim3(256), 0, st>>>(
        A, lda, aoff, W, A2, lda2, W2, b1, b2, (bf16_t*)C, ldc, M, Nc, K, K2);
  }
}

#define EW(n) dim3(((n) + 255) / 256), dim3(256), 0, stream

extern "C" void kernel_launch(void* const* d_in, const int* in_sizes, int n_in,
                              void* d_out, int out_size, void* d_ws, size_t ws_size,
                              hipStream_t stream) {
  (void)in_sizes; (void)n_in; (void)out_size; (void)ws_size;
  const float* x     = (const float*)d_in[0];
  const int*   ei    = (const int*)d_in[1];
  const float* w_in0 = (const float*)d_in[2];
  const float* b_in0 = (const float*)d_in[3];
  const float* w_in1 = (const float*)d_in[4];
  const float* b_in1 = (const float*)d_in[5];
  const float* wih[2] = {(const float*)d_in[6],  (const float*)d_in[10]};
  const float* whh[2] = {(const float*)d_in[7],  (const float*)d_in[11]};
  const float* bih[2] = {(const float*)d_in[8],  (const float*)d_in[12]};
  const float* bhh[2] = {(const float*)d_in[9],  (const float*)d_in[13]};
  const float* g_wl[2]   = {(const float*)d_in[14], (const float*)d_in[22]};
  const float* g_bl[2]   = {(const float*)d_in[15], (const float*)d_in[23]};
  const float* g_wr[2]   = {(const float*)d_in[16], (const float*)d_in[24]};
  const float* g_br[2]   = {(const float*)d_in[17], (const float*)d_in[25]};
  const float* g_att[2]  = {(const float*)d_in[18], (const float*)d_in[26]};
  const float* g_bias[2] = {(const float*)d_in[19], (const float*)d_in[27]};
  const float* g_pw[2]   = {(const float*)d_in[20], (const float*)d_in[28]};
  const float* g_pb[2]   = {(const float*)d_in[21], (const float*)d_in[29]};
  const float* w_out0 = (const float*)d_in[30];
  const float* b_out0 = (const float*)d_in[31];
  const float* w_out1 = (const float*)d_in[32];
  const float* b_out1 = (const float*)d_in[33];
  float* out = (float*)d_out;

  // ---- workspace carve-out (≈175 MB) ----
  char* ws = (char*)d_ws;
  size_t cur = 0;
  auto alloc = [&](size_t bytes) -> void* {
    cur = (cur + 255) & ~(size_t)255;
    void* p = ws + cur;
    cur += bytes;
    return p;
  };
  const long NT = (long)NND * TT;                           // 240000 rows
  bf16_t* xpad = (bf16_t*)alloc(NT * 32 * 2);               // padded input; reused: score f32
  bf16_t* bufA = (bf16_t*)alloc(NT * HIDD * 2);             // mlp hidden1 -> hseq0 -> xl/xr/agg
  bf16_t* bufB = (bf16_t*)alloc(NT * HIDD * 2);             // mlp out (lstm0 input) -> zcat
  bf16_t* w0p  = (bf16_t*)alloc(128 * 32 * 2);
  bf16_t* w1b  = (bf16_t*)alloc(128 * 128 * 2);
  bf16_t* wihb[2], *whhb[2], *wlb[2], *wrb[2], *pwb[2];
  for (int l = 0; l < 2; ++l) {
    wihb[l] = (bf16_t*)alloc(512 * 128 * 2);
    whhb[l] = (bf16_t*)alloc(512 * 128 * 2);
  }
  for (int l = 0; l < 2; ++l) {
    wlb[l] = (bf16_t*)alloc(512 * 128 * 2);
    wrb[l] = (bf16_t*)alloc(512 * 128 * 2);
    pwb[l] = (bf16_t*)alloc(128 * 512 * 2);
  }
  bf16_t* wo0b = (bf16_t*)alloc(128 * 128 * 2);
  bf16_t* wo1p = (bf16_t*)alloc(64 * 128 * 2);              // 24x128 padded to 64 rows
  bf16_t* h_cur = (bf16_t*)alloc((size_t)NND * HIDD * 2);
  float*  cbuf  = (float*)alloc((size_t)NND * HIDD * 4);
  float*  gates = (float*)alloc((size_t)NND * 512 * 4);
  float*  smax  = (float*)alloc((size_t)NND * NHEADS * 4);
  float*  denom = (float*)alloc((size_t)NND * NHEADS * 4);
  bf16_t* zbuf  = (bf16_t*)alloc((size_t)NND * HIDD * 2);
  bf16_t* m1    = (bf16_t*)alloc((size_t)NND * HIDD * 2);
  // aliases (live only after the LSTM / before GAT consumers)
  float* xl  = (float*)bufA;                                 // 10000x512 f32
  float* xr  = xl + (long)NND * 512;
  float* agg = xr + (long)NND * 512;
  float* score = (float*)xpad;                               // E*4 f32
  bf16_t* zcat = bufB;                                       // 10000x512 bf16

  // ---- cast weights / input to bf16 ----
  cast_pad2d_kernel<<<EW(NT * 32)>>>(x, xpad, (int)NT, IND, (int)NT, 32);
  cast_pad2d_kernel<<<EW(128 * 32)>>>(w_in0, w0p, 128, 16, 128, 32);
  cast_kernel<<<EW(128 * 128)>>>(w_in1, w1b, 128 * 128);
  for (int l = 0; l < 2; ++l) {
    cast_kernel<<<EW(512 * 128)>>>(wih[l], wihb[l], 512 * 128);
    cast_kernel<<<EW(512 * 128)>>>(whh[l], whhb[l], 512 * 128);
    cast_kernel<<<EW(512 * 128)>>>(g_wl[l], wlb[l], 512 * 128);
    cast_kernel<<<EW(512 * 128)>>>(g_wr[l], wrb[l], 512 * 128);
    cast_kernel<<<EW(128 * 512)>>>(g_pw[l], pwb[l], 128 * 512);
  }
  cast_kernel<<<EW(128 * 128)>>>(w_out0, wo0b, 128 * 128);
  cast_pad2d_kernel<<<EW(64 * 128)>>>(w_out1, wo1p, PREDL, 128, 64, 128);

  // ---- input MLP: [240000,32]@[128,32]^T -> bufA; @[128,128]^T+relu -> bufB ----
  launch_gemm(stream, 1, 0, xpad, 32, 0, w0p, nullptr, 0, nullptr, b_in0, nullptr,
              bufA, HIDD, (int)NT, HIDD, 32, 0);
  launch_gemm(stream, 1, 1, bufA, HIDD, 0, w1b, nullptr, 0, nullptr, b_in1, nullptr,
              bufB, HIDD, (int)NT, HIDD, HIDD, 0);

  // ---- LSTM (2 layers, sequential over T); gate GEMM fuses x_t*Wih^T + h*Whh^T ----
  for (int l = 0; l < 2; ++l) {
    hipMemsetAsync(h_cur, 0, (size_t)NND * HIDD * 2, stream);
    hipMemsetAsync(cbuf, 0, (size_t)NND * HIDD * 4, stream);
    const bf16_t* seq_in = (l == 0) ? bufB : bufA;
    bf16_t* seq_out = (l == 0) ? bufA : nullptr;             // layer1: only final h needed
    for (int t = 0; t < TT; ++t) {
      launch_gemm(stream, 0, 0, seq_in, (long)TT * HIDD, (long)t * HIDD, wihb[l],
                  h_cur, HIDD, whhb[l], bih[l], bhh[l],
                  gates, 4 * HIDD, NND, 4 * HIDD, HIDD, HIDD);
      lstm_cell_kernel<<<EW(NND * HIDD)>>>(gates, cbuf, h_cur, seq_out, t);
    }
  }
  // z0 = h_cur (bf16 [10000,128])

  // ---- 2x (GATv2 -> Linear(512,128)+ReLU) ----
  const bf16_t* zin = h_cur;
  for (int l = 0; l < 2; ++l) {
    launch_gemm(stream, 0, 0, zin, HIDD, 0, wlb[l], nullptr, 0, nullptr, g_bl[l], nullptr,
                xl, 512, NND, 512, HIDD, 0);
    launch_gemm(stream, 0, 0, zin, HIDD, 0, wrb[l], nullptr, 0, nullptr, g_br[l], nullptr,
                xr, 512, NND, 512, HIDD, 0);
    hipMemsetAsync(agg, 0, (size_t)NND * 512 * 4, stream);
    hipMemsetAsync(denom, 0, (size_t)NND * NHEADS * 4, stream);
    fill_kernel<<<EW(NND * NHEADS)>>>(smax, -INFINITY, NND * NHEADS);
    gat_score_kernel<<<EW(NE * NHEADS)>>>(ei, xl, xr, g_att[l], score, smax);
    gat_norm_kernel<<<EW(NE * NHEADS)>>>(ei, score, smax, denom);
    {
      long tot = (long)NE * 512;
      gat_agg_kernel<<<dim3((unsigned)((tot + 255) / 256)), dim3(256), 0, stream>>>(
          ei, score, denom, xl, agg);
    }
    addbias_cast_kernel<<<EW(NND * 512)>>>(agg, g_bias[l], zcat, NND * 512);
    bf16_t* zo = (l == 0) ? zbuf : h_cur;                    // layer1 may reuse h_cur
    launch_gemm(stream, 1, 1, zcat, 512, 0, pwb[l], nullptr, 0, nullptr, g_pb[l], nullptr,
                zo, HIDD, NND, HIDD, 512, 0);
    zin = zo;
  }

  // ---- output MLP: relu(z@wo0^T+b0) -> m1; m1@wo1^T+b1 -> out [10000,24] ----
  launch_gemm(stream, 1, 1, zin, HIDD, 0, wo0b, nullptr, 0, nullptr, b_out0, nullptr,
              m1, HIDD, NND, HIDD, HIDD, 0);
  launch_gemm(stream, 0, 0, m1, HIDD, 0, wo1p, nullptr, 0, nullptr, b_out1, nullptr,
              out, PREDL, NND, PREDL, HIDD, 0);
}
